// ExplodedLogit_30262339567618
// MI455X (gfx1250) — matrix operand order repaired and checked by reading, not compiled, not run on hardware
//
#include <hip/hip_runtime.h>

typedef _Float16 v16h __attribute__((ext_vector_type(16)));
typedef float    v8f  __attribute__((ext_vector_type(8)));
typedef float    f32x4 __attribute__((ext_vector_type(4)));

#define TRACKS  512
#define FEATS   256
#define ROW_LEN 262145              // 1 + 512*512
#define MASK_VAL (-105.9189148f)    // log(1e-46) in fp32

// ---------------------------------------------------------------------------
// Kernel 1: scores = x @ W.T + b via v_wmma_f32_16x16x32_f16.
// One wave per 16-row tile. B has all 16 columns equal to the W k-chunk, so
// every column of D holds the 16 scores; lane 0 / lane 16 own them in C/D
// VGPRs 0..7 per the 16x16 f32 C/D layout.
// ---------------------------------------------------------------------------
__global__ __launch_bounds__(32)
void el_scores_wmma(const float* __restrict__ x, const float* __restrict__ W,
                    const float* __restrict__ b, float* __restrict__ scores) {
    const int lane = threadIdx.x;           // 0..31, one wave32 per block
    const int tile = blockIdx.x;            // 0..31 (32 tiles of 16 rows)
    const int mrow = tile * 16 + (lane & 15);
    const int kloA = (lane < 16) ? 0 : 8;   // A layout: lo lanes K0-7/16-23, hi lanes K8-15/24-31
    const int kloB = (lane < 16) ? 0 : 8;   // B layout: lo lanes K0-7/16-23, hi lanes K8-15/24-31

    const float* __restrict__ xr = x + mrow * FEATS;
    v8f c = {};
#pragma unroll
    for (int kb = 0; kb < FEATS; kb += 32) {
        v16h a, bb;
#pragma unroll
        for (int t = 0; t < 8; ++t) {
            a[t]     = (_Float16)xr[kb + kloA + t];
            a[8 + t] = (_Float16)xr[kb + 16 + kloA + t];
        }
#pragma unroll
        for (int t = 0; t < 16; ++t)
            bb[t] = (_Float16)W[kb + kloB + t + (t & 8)];   // every column of B == W chunk
        c = __builtin_amdgcn_wmma_f32_16x16x32_f16(
                /*neg_a=*/false, a, /*neg_b=*/false, bb,
                /*c_mod=*/(short)0, c, /*reuse_a=*/false, /*reuse_b=*/false);
    }
    const float bias = b[0];
    if (lane == 0) {                 // D[M=r][N=0], r = 0..7
#pragma unroll
        for (int r = 0; r < 8; ++r) scores[tile * 16 + r] = c[r] + bias;
    } else if (lane == 16) {         // D[M=r+8][N=0], r = 0..7
#pragma unroll
        for (int r = 0; r < 8; ++r) scores[tile * 16 + 8 + r] = c[r] + bias;
    }
}

// ---------------------------------------------------------------------------
// Kernel 2: first-occurrence argmax over the 512 scores (matches jnp.argmax).
// ---------------------------------------------------------------------------
__global__ __launch_bounds__(512)
void el_argmax(const float* __restrict__ scores, int* __restrict__ idx_out) {
    __shared__ float sv[512];
    __shared__ int   si[512];
    const int t = threadIdx.x;
    sv[t] = scores[t];
    si[t] = t;
    __syncthreads();
    for (int s = 256; s > 0; s >>= 1) {
        if (t < s) {
            float v1 = sv[t], v2 = sv[t + s];
            int   i1 = si[t], i2 = si[t + s];
            if (v2 > v1 || (v2 == v1 && i2 < i1)) { sv[t] = v2; si[t] = i2; }
        }
        __syncthreads();
    }
    if (t == 0) idx_out[0] = si[0];
}

// ---------------------------------------------------------------------------
// Kernel 3: stream 537 MB of output. Row i = [score_i, 512 x (pattern of 512)]
// where pattern[j] = score_i, except pattern[idx] = score_i * MASK_VAL.
// 16B-aligned b128 non-temporal stores; odd row length handled via per-row
// realignment with <=3-element scalar head/tail.
// ---------------------------------------------------------------------------
__global__ __launch_bounds__(256)
void el_fill(const float* __restrict__ scoresWs, const int* __restrict__ idxWs,
             float* __restrict__ out) {
    const int      row = blockIdx.y;
    const float    v   = scoresWs[row];
    const unsigned idx = (unsigned)idxWs[0];
    const float    mv  = v * MASK_VAL;

    const size_t   base = (size_t)row * ROW_LEN;   // col 0 (score)
    const size_t   s    = base + 1;                // interior start (col 1)
    const size_t   a0   = (s + 3) & ~(size_t)3;    // first 16B-aligned element
    const unsigned h    = (unsigned)(a0 - s);      // head elements (0..3)
    const unsigned total4 = (262144u - h) >> 2;    // aligned float4 count

    const unsigned gbeg = blockIdx.x * 1024u;
    const unsigned gend = (gbeg + 1024u < total4) ? gbeg + 1024u : total4;
    unsigned g = gbeg + threadIdx.x;
#pragma unroll
    for (int u = 0; u < 4; ++u, g += 256u) {
        if (g < gend) {
            const size_t   q  = a0 + ((size_t)g << 2);
            const unsigned c1 = (unsigned)(q - s);   // (col-1) of first lane elem
            f32x4 val;
#pragma unroll
            for (int e = 0; e < 4; ++e) {
                const unsigned j = (c1 + (unsigned)e) & 511u;
                val[e] = (j == idx) ? mv : v;
            }
            __builtin_nontemporal_store(val, (f32x4*)(out + q));
        }
    }

    if (blockIdx.x == 0 && threadIdx.x == 0) {
        out[base] = v;                                   // col 0: the score
        for (unsigned e = 0; e < h; ++e)                 // unaligned head
            out[s + e] = ((e & 511u) == idx) ? mv : v;
        const unsigned tail = (4u - h) & 3u;             // unaligned tail
        const size_t   tq   = a0 + ((size_t)total4 << 2);
        for (unsigned e = 0; e < tail; ++e) {
            const unsigned j = (unsigned)(tq + e - s) & 511u;
            out[tq + e] = (j == idx) ? mv : v;
        }
    }
}

extern "C" void kernel_launch(void* const* d_in, const int* in_sizes, int n_in,
                              void* d_out, int out_size, void* d_ws, size_t ws_size,
                              hipStream_t stream) {
    const float* x = (const float*)d_in[0];   // (512, 256)
    const float* W = (const float*)d_in[1];   // (1, 256)
    const float* b = (const float*)d_in[2];   // (1,)
    float* out = (float*)d_out;               // (512, 262145) flat

    float* wsf = (float*)d_ws;                // ws[0..511]  : scores
    int*   wsi = (int*)(wsf + TRACKS);        // ws[512]     : argmax index

    el_scores_wmma<<<dim3(32), dim3(32), 0, stream>>>(x, W, b, wsf);
    el_argmax<<<dim3(1), dim3(512), 0, stream>>>(wsf, wsi);
    el_fill<<<dim3(64, 512), dim3(256), 0, stream>>>(wsf, wsi, out);
}
